// VQVAEBase_5231270166659
// MI455X (gfx1250) — matrix operand order repaired and checked by reading, not compiled, not run on hardware
//
#include <hip/hip_runtime.h>
#include <stdint.h>
#include <stddef.h>

// ---------------------------------------------------------------------------
// CDNA5 (gfx1250) VQ-VAE forward. Heavy convs = implicit-GEMM with
// v_wmma_f32_16x16x32_bf16 (wave32). Weight tiles staged global->LDS with
// gfx1250 async loads (global_load_async_to_lds_b128 + s_wait_asynccnt).
// Activations im2col-gathered into LDS. Small convs / VQ / losses scalar.
// ---------------------------------------------------------------------------

typedef __attribute__((ext_vector_type(16))) __bf16 v16bf;
typedef __attribute__((ext_vector_type(8)))  float  v8f;

__device__ __forceinline__ uint16_t f2bf(float f) {
  union { float f; uint32_t u; } c; c.f = f;
  uint32_t u = c.u;
  uint32_t r = ((u >> 16) & 1u) + 0x7FFFu;   // round-to-nearest-even
  return (uint16_t)((u + r) >> 16);
}

// --------------------------- elementwise: xn = (x-128)/128 ------------------
__global__ void k_xn(const float* __restrict__ x, float* __restrict__ xn, int n) {
  int i = blockIdx.x * 256 + threadIdx.x;
  if (i < n) xn[i] = (x[i] - 128.0f) * (1.0f / 128.0f);
}

// --------------------------- e_c1: 3->256, k4 s2 p1, 64->32 (scalar) --------
__global__ void k_conv_ec1(const float* __restrict__ xn, const float* __restrict__ w,
                           const float* __restrict__ bias, float* __restrict__ out) {
  int gid = blockIdx.x * 256 + threadIdx.x;          // ((b*256+co)*32+oy)*32+ox
  int ox = gid & 31;
  int oy = (gid >> 5) & 31;
  int co = (gid >> 10) & 255;
  int b  = gid >> 18;
  float acc = bias[co];
  for (int ci = 0; ci < 3; ++ci)
    for (int ky = 0; ky < 4; ++ky) {
      int iy = oy * 2 - 1 + ky;
      if (iy < 0 || iy >= 64) continue;
      for (int kx = 0; kx < 4; ++kx) {
        int ix = ox * 2 - 1 + kx;
        if (ix < 0 || ix >= 64) continue;
        acc += xn[((b * 3 + ci) * 64 + iy) * 64 + ix] *
               w[((co * 3 + ci) * 4 + ky) * 4 + kx];
      }
    }
  out[gid] = acc;
}

// ------------------- weight packing to bf16, layout [Cout][Kpad] ------------
// (N-major so a 128x32 B-tile is 128 contiguous 64B rows -> async b128 copy)
__global__ void k_pack_w(const float* __restrict__ w, uint16_t* __restrict__ wpk,
                         int Cin, int Cout, int kh, int kw, int Kpad, int transposed) {
  int gid = blockIdx.x * 256 + threadIdx.x;
  if (gid >= Kpad * Cout) return;
  int co = gid % Cout;
  int k  = gid / Cout;
  int K  = Cin * kh * kw;
  float v = 0.0f;
  if (k < K) {
    int ci = k / (kh * kw);
    int rr = k - ci * kh * kw;
    int ky = rr / kw, kx = rr - ky * kw;
    v = transposed ? w[((ci * Cout + co) * kh + ky) * kw + kx]    // (Cin,Cout,kh,kw)
                   : w[((co * Cin + ci) * kh + ky) * kw + kx];    // (Cout,Cin,kh,kw)
  }
  wpk[(size_t)co * Kpad + k] = f2bf(v);
}

// --------------------------- WMMA implicit-GEMM conv ------------------------
// out[m,n] = sum_k A[m,k]*B[k,n] + bias[n];  m=(b,oy,ox), n=co, k=(ci,ky,kx)
// Block: 256 threads = 8 waves (4 M-waves x 2 N-waves) -> 64(M) x 128(N) tile.
// Each wave: 16x64 via 4 WMMA per 32-wide K-step (A fragment reused 4x).
__global__ void k_conv_wmma(const uint16_t* __restrict__ act,
                            const uint16_t* __restrict__ wpk,   // [Cout][Kpad]
                            const float* __restrict__ bias, float* __restrict__ out,
                            int Cin, int Hin, int Win, int Cout, int Hout, int Wout,
                            int kh, int kw, int stride, int pad, int Kpad, int transposed) {
  __shared__ __align__(16) uint16_t Atile[64 * 32];      // [M][K] row-major
  __shared__ __align__(16) uint16_t BtileT[128 * 32];    // [N][K] row-major

  const int tid  = threadIdx.x;
  const int lane = tid & 31;
  const int wave = tid >> 5;
  const int wm   = wave & 3;       // M quarter (0..3)
  const int wn   = wave >> 2;      // N half    (0..1)
  const int r    = lane & 15;
  const int hi   = lane >> 4;

  const int m0  = blockIdx.x * 64;
  const int n0  = blockIdx.y * 128;
  const int K   = Cin * kh * kw;
  const int khw = kh * kw;
  const int HWo = Hout * Wout;

  // Per-thread A-staging slots: m-decomposition is K-invariant -> hoist.
  int pb[8], poy[8], pox[8];
  #pragma unroll
  for (int t = 0; t < 8; ++t) {
    int mg = m0 + ((tid + t * 256) >> 5);
    int b  = mg / HWo;
    int hw = mg - b * HWo;
    int oy = hw / Wout;
    pb[t] = b; poy[t] = oy; pox[t] = hw - oy * Wout;
  }
  // Per-thread async B-copy slots (two 16B chunks): row nn, chunk ch.
  const int nn0 = tid >> 2,        ch0 = tid & 3;         // e = tid
  const int nn1 = (tid + 256) >> 2, ch1 = (tid + 256) & 3; // e = tid+256

  v8f acc0 = {}, acc1 = {}, acc2 = {}, acc3 = {};

  for (int k0 = 0; k0 < Kpad; k0 += 32) {
    __syncthreads();
    // ---- B tile: async DMA global->LDS (128 rows x 64B), overlaps A gather
    {
      uint64_t gb = (uint64_t)(uintptr_t)wpk + ((size_t)n0 * Kpad + k0) * 2;
      uint32_t lb = (uint32_t)(uintptr_t)&BtileT[0];
      uint64_t g0 = gb + (size_t)nn0 * Kpad * 2 + ch0 * 16;
      uint32_t l0 = lb + nn0 * 64 + ch0 * 16;
      uint64_t g1 = gb + (size_t)nn1 * Kpad * 2 + ch1 * 16;
      uint32_t l1 = lb + nn1 * 64 + ch1 * 16;
      asm volatile("global_load_async_to_lds_b128 %0, %1, off"
                   :: "v"(l0), "v"(g0) : "memory");
      asm volatile("global_load_async_to_lds_b128 %0, %1, off"
                   :: "v"(l1), "v"(g1) : "memory");
    }
    // ---- stage A (64x32 bf16) via on-the-fly im2col gather ----
    #pragma unroll
    for (int t = 0; t < 8; ++t) {
      int e    = tid + t * 256;
      int mrow = e >> 5;
      int kk   = e & 31;
      int kg   = k0 + kk;
      uint16_t v = 0;
      if (kg < K) {
        int ci = kg / khw;
        int rr = kg - ci * khw;
        int ky = rr / kw, kx = rr - ky * kw;
        int oy = poy[t], ox = pox[t];
        int iy, ix; bool ok;
        if (!transposed) {
          iy = oy * stride - pad + ky;
          ix = ox * stride - pad + kx;
          ok = (iy >= 0 && iy < Hin && ix >= 0 && ix < Win);
        } else {                      // conv-transpose, stride 2: oy = 2*iy - pad + ky
          int ty = oy + pad - ky, tx = ox + pad - kx;
          ok = ((ty & 1) == 0) && ((tx & 1) == 0) && ty >= 0 && tx >= 0;
          iy = ty >> 1; ix = tx >> 1;
          ok = ok && iy < Hin && ix < Win;
        }
        if (ok) v = act[((size_t)(pb[t] * Cin + ci) * Hin + iy) * Win + ix];
      }
      Atile[mrow * 32 + kk] = v;
    }
    if (k0 + 32 < Kpad)   // gfx1250 global_prefetch_b8 on next weight K-tile
      __builtin_prefetch(wpk + (size_t)(n0 + (tid & 127)) * Kpad + k0 + 32, 0, 1);

    asm volatile("s_wait_asynccnt 0x0" ::: "memory");   // B DMA complete (this wave)
    __syncthreads();

    // ---- load fragments: contiguous k order (k = 16*hi + s), same for A and B ----
    union { v16bf v; uint32_t d[8]; } af, b0, b1, b2, b3;
    const uint32_t* Arow = (const uint32_t*)&Atile[(wm * 16 + r) * 32];  // 16 dwords/row
    #pragma unroll
    for (int j = 0; j < 8; ++j) {      // lanes<16: K0..15; lanes>=16: K16..31
      af.d[j] = Arow[hi * 8 + j];
    }
    const uint32_t* Bb = (const uint32_t*)BtileT + (size_t)(wn * 64 + r) * 16;
    #pragma unroll
    for (int j = 0; j < 8; ++j) {      // lanes<16: K0..15; lanes>=16: K16..31
      b0.d[j] = Bb[hi * 8 + j];
      b1.d[j] = Bb[256 + hi * 8 + j];   // +16 columns
      b2.d[j] = Bb[512 + hi * 8 + j];
      b3.d[j] = Bb[768 + hi * 8 + j];
    }
    acc0 = __builtin_amdgcn_wmma_f32_16x16x32_bf16(false, af.v, false, b0.v, (short)0, acc0, false, false);
    acc1 = __builtin_amdgcn_wmma_f32_16x16x32_bf16(false, af.v, false, b1.v, (short)0, acc1, false, false);
    acc2 = __builtin_amdgcn_wmma_f32_16x16x32_bf16(false, af.v, false, b2.v, (short)0, acc2, false, false);
    acc3 = __builtin_amdgcn_wmma_f32_16x16x32_bf16(false, af.v, false, b3.v, (short)0, acc3, false, false);
  }

  // ---- epilogue: C/D layout -> NCHW f32 + bias ----
  const int nb = n0 + wn * 64 + r;
  #pragma unroll
  for (int j = 0; j < 8; ++j) {
    int mg = m0 + wm * 16 + hi * 8 + j;
    int b  = mg / HWo;
    int hw = mg - b * HWo;
    int oy = hw / Wout, ox = hw - oy * Wout;
    size_t base = ((size_t)b * Cout) * HWo + (size_t)oy * Wout + ox;
    out[base + (size_t)(nb +  0) * HWo] = acc0[j] + bias[nb +  0];
    out[base + (size_t)(nb + 16) * HWo] = acc1[j] + bias[nb + 16];
    out[base + (size_t)(nb + 32) * HWo] = acc2[j] + bias[nb + 32];
    out[base + (size_t)(nb + 48) * HWo] = acc3[j] + bias[nb + 48];
  }
}

// ------------- LayerNorm(channel, C=256) + opt residual + ReLU --------------
__global__ void k_ln(const float* __restrict__ in, const float* __restrict__ g,
                     const float* __restrict__ bta, const float* __restrict__ res,
                     float* __restrict__ outf, uint16_t* __restrict__ outh, int HW) {
  const int p  = blockIdx.x;           // pixel index in [0, B*HW)
  const int c  = threadIdx.x;          // channel 0..255
  const int b  = p / HW;
  const int hw = p - b * HW;
  const size_t idx = ((size_t)(b * 256 + c)) * HW + hw;
  const float v = in[idx];

  float s = v, s2 = v * v;
  #pragma unroll
  for (int o = 16; o > 0; o >>= 1) {
    s  += __shfl_down(s,  o, 32);
    s2 += __shfl_down(s2, o, 32);
  }
  __shared__ float ws0[8], ws1[8];
  const int lane = c & 31, w = c >> 5;
  if (lane == 0) { ws0[w] = s; ws1[w] = s2; }
  __syncthreads();
  if (w == 0) {
    float a  = (lane < 8) ? ws0[lane] : 0.0f;
    float a2 = (lane < 8) ? ws1[lane] : 0.0f;
    #pragma unroll
    for (int o = 4; o > 0; o >>= 1) {
      a  += __shfl_down(a,  o, 32);
      a2 += __shfl_down(a2, o, 32);
    }
    if (lane == 0) { ws0[0] = a; ws1[0] = a2; }
  }
  __syncthreads();
  const float mu   = ws0[0] * (1.0f / 256.0f);
  const float var  = ws1[0] * (1.0f / 256.0f) - mu * mu;
  const float rstd = rsqrtf(var + 1e-5f);
  float y = (v - mu) * rstd * g[c] + bta[c];
  if (res) y += res[idx];
  y = fmaxf(y, 0.0f);
  if (outf) outf[idx] = y;
  if (outh) outh[idx] = f2bf(y);
}

// --------------------------- VQ: nearest code per pixel ---------------------
__global__ void k_vq_assign(const float* __restrict__ enc, const float* __restrict__ cb,
                            int* __restrict__ idx, int HW) {
  const int p = blockIdx.x;            // pixel
  const int t = threadIdx.x;           // code 0..127
  __shared__ float e[256];
  const int b = p / HW, hw = p - b * HW;
  e[t]       = enc[((size_t)(b * 256 + t)) * HW + hw];
  e[t + 128] = enc[((size_t)(b * 256 + t + 128)) * HW + hw];
  __syncthreads();
  float d2 = 0.0f;
  const float* c = cb + (size_t)t * 256;
  for (int d = 0; d < 256; ++d) { float df = e[d] - c[d]; d2 += df * df; }
  __shared__ float vals[128];
  __shared__ int   is[128];
  vals[t] = d2; is[t] = t;
  __syncthreads();
  for (int o = 64; o > 0; o >>= 1) {
    if (t < o) {
      float a = vals[t], bb = vals[t + o];
      if (bb < a || (bb == a && is[t + o] < is[t])) { vals[t] = bb; is[t] = is[t + o]; }
    }
    __syncthreads();
  }
  if (t == 0) idx[p] = is[0];
}

__global__ void k_vq_gather(const float* __restrict__ cb, const int* __restrict__ idx,
                            float* __restrict__ embf, float* __restrict__ s1,
                            uint16_t* __restrict__ embh, int HW) {
  int gid = blockIdx.x * 256 + threadIdx.x;   // p*256 + d
  int d = gid & 255, p = gid >> 8;
  int b = p / HW, hw = p - b * HW;
  float v = cb[(size_t)idx[p] * 256 + d];
  size_t o = ((size_t)(b * 256 + d)) * HW + hw;
  embf[o] = v; s1[o] = v; embh[o] = f2bf(v);
}

// ------------- d_dc2: 256->3 conv-transpose k4 s2 p1, 32->64, +tanh ---------
__global__ void k_deconv2_tanh(const float* __restrict__ in, const float* __restrict__ w,
                               const float* __restrict__ bias, float* __restrict__ recon) {
  int gid = blockIdx.x * 256 + threadIdx.x;   // ((b*3+co)*64+oy)*64+ox
  int ox = gid & 63;
  int oy = (gid >> 6) & 63;
  int t  = gid >> 12;
  int co = t % 3;
  int b  = t / 3;
  float acc = bias[co];
  for (int ky = 0; ky < 4; ++ky) {
    int ty = oy + 1 - ky;
    if (ty & 1) continue;
    int iy = ty >> 1;
    if (iy < 0 || iy >= 32) continue;
    for (int kx = 0; kx < 4; ++kx) {
      int tx = ox + 1 - kx;
      if (tx & 1) continue;
      int ix = tx >> 1;
      if (ix < 0 || ix >= 32) continue;
      const float* ip = in + (size_t)b * 256 * 1024 + iy * 32 + ix;
      const float* wp = w + co * 16 + ky * 4 + kx;     // w[(ci*3+co)*16 + ky*4+kx]
      for (int ci = 0; ci < 256; ++ci) acc += ip[(size_t)ci * 1024] * wp[ci * 48];
    }
  }
  recon[gid] = tanhf(acc);
}

// --------------------------- per-batch MSE ----------------------------------
__global__ void k_loss(const float* __restrict__ a, const float* __restrict__ b,
                       int n, float* __restrict__ o1, float* __restrict__ o2) {
  const int bb = blockIdx.x, t = threadIdx.x;
  const float* pa = a + (size_t)bb * n;
  const float* pb = b + (size_t)bb * n;
  float s = 0.0f;
  for (int i = t; i < n; i += 256) { float d = pa[i] - pb[i]; s += d * d; }
  #pragma unroll
  for (int o = 16; o > 0; o >>= 1) s += __shfl_down(s, o, 32);
  __shared__ float ws0[8];
  if ((t & 31) == 0) ws0[t >> 5] = s;
  __syncthreads();
  if (t == 0) {
    float tot = 0.0f;
    for (int i = 0; i < 8; ++i) tot += ws0[i];
    tot /= (float)n;
    o1[bb] = tot;
    if (o2) o2[bb] = tot;
  }
}

__global__ void k_finalize(float* __restrict__ out) {
  int b = threadIdx.x;   // 16 threads
  out[b] = out[16 + b] + out[32 + b] + 0.25f * out[48 + b];
}

// ===========================================================================
extern "C" void kernel_launch(void* const* d_in, const int* in_sizes, int n_in,
                              void* d_out, int out_size, void* d_ws, size_t ws_size,
                              hipStream_t stream) {
  (void)in_sizes; (void)n_in; (void)out_size; (void)ws_size;
  auto P = [&](int i) { return (const float*)d_in[i]; };
  // setup_inputs() dict order: x, then params insertion order.
  const float* x      = P(0);
  const float* e_c1w  = P(1);  const float* e_c1b  = P(2);
  const float* e_n1g  = P(3);  const float* e_n1b  = P(4);
  const float* e_c2w  = P(5);  const float* e_c2b  = P(6);
  const float* e_n2g  = P(7);  const float* e_n2b  = P(8);
  const float* e_c3w  = P(9);  const float* e_c3b  = P(10);
  const float* d_dc1w = P(11); const float* d_dc1b = P(12);
  const float* d_n1g  = P(13); const float* d_n1b  = P(14);
  const float* d_dc2w = P(15); const float* d_dc2b = P(16);
  const float* cbk    = P(17);
  // resblocks: e_r1=18, e_r2=26, d_r1=34, d_r2=42; within: c1w,c1b,n1g,n1b,c2w,c2b,n2g,n2b

  // ----- workspace carve (bytes) -----
  char* ws = (char*)d_ws;
  size_t off = 0;
  auto carve = [&](size_t bytes) { void* p = ws + off; off += (bytes + 255) & ~(size_t)255; return p; };
  float*    XN    = (float*)carve(196608 * 4);
  float*    RECON = (float*)carve(196608 * 4);
  int*      IDX   = (int*)carve(4096 * 4);
  float*    BIGA  = (float*)carve(4194304 * 4);     // (16,256,32,32) f32
  void*     BIGB  = carve(4194304 * 4);             // bf16 a1, later f32 LN out
  float*    S0    = (float*)carve(1048576 * 4);
  float*    S1    = (float*)carve(1048576 * 4);
  float*    S2    = (float*)carve(1048576 * 4);
  float*    S3    = (float*)carve(1048576 * 4);     // encoded
  float*    S4    = (float*)carve(1048576 * 4);     // emb
  uint16_t* HB0   = (uint16_t*)carve(1048576 * 2);
  uint16_t* HB1   = (uint16_t*)carve(1048576 * 2);
  uint16_t* WPK   = (uint16_t*)carve(4096 * 256 * 2);
  float*    out   = (float*)d_out;

  // 1) normalize input
  k_xn<<<768, 256, 0, stream>>>(x, XN, 196608);
  // 2) e_c1 (scalar) -> BIGA (16,256,32,32)
  k_conv_ec1<<<16384, 256, 0, stream>>>(XN, e_c1w, e_c1b, BIGA);
  // 3) LN1 + ReLU -> bf16 BIGB
  k_ln<<<16384, 256, 0, stream>>>(BIGA, e_n1g, e_n1b, nullptr, nullptr, (uint16_t*)BIGB, 1024);
  // 4) e_c2: WMMA conv 256->256, k4 s2 p1, 32->16  (M=4096)
  k_pack_w<<<4096, 256, 0, stream>>>(e_c2w, WPK, 256, 256, 4, 4, 4096, 0);
  k_conv_wmma<<<dim3(64, 2), 256, 0, stream>>>((uint16_t*)BIGB, WPK, e_c2b, S0,
      256, 32, 32, 256, 16, 16, 4, 4, 2, 1, 4096, 0);
  // 5) LN2 + ReLU -> S1 (f32 residual base) + HB0 (bf16)
  k_ln<<<4096, 256, 0, stream>>>(S0, e_n2g, e_n2b, nullptr, S1, HB0, 256);

  // resblock: (S1,HB0) -> (S1,HB0); scratch S2,S0,HB1,WPK
  auto resblock = [&](int base) {
    k_pack_w<<<2304, 256, 0, stream>>>(P(base + 0), WPK, 256, 256, 3, 3, 2304, 0);
    k_conv_wmma<<<dim3(64, 2), 256, 0, stream>>>(HB0, WPK, P(base + 1), S2,
        256, 16, 16, 256, 16, 16, 3, 3, 1, 1, 2304, 0);
    k_ln<<<4096, 256, 0, stream>>>(S2, P(base + 2), P(base + 3), nullptr, nullptr, HB1, 256);
    k_pack_w<<<2304, 256, 0, stream>>>(P(base + 4), WPK, 256, 256, 3, 3, 2304, 0);
    k_conv_wmma<<<dim3(64, 2), 256, 0, stream>>>(HB1, WPK, P(base + 5), S0,
        256, 16, 16, 256, 16, 16, 3, 3, 1, 1, 2304, 0);
    k_ln<<<4096, 256, 0, stream>>>(S0, P(base + 6), P(base + 7), S1, S1, HB0, 256);
  };
  resblock(18);   // e_r1
  resblock(26);   // e_r2

  // 6) e_c3 1x1 -> encoded S3
  k_pack_w<<<256, 256, 0, stream>>>(e_c3w, WPK, 256, 256, 1, 1, 256, 0);
  k_conv_wmma<<<dim3(64, 2), 256, 0, stream>>>(HB0, WPK, e_c3b, S3,
      256, 16, 16, 256, 16, 16, 1, 1, 1, 0, 256, 0);

  // 7) VQ
  k_vq_assign<<<4096, 128, 0, stream>>>(S3, cbk, IDX, 256);
  k_vq_gather<<<4096, 256, 0, stream>>>(cbk, IDX, S4, S1, HB0, 256);  // st == emb forward

  // 8) decoder resblocks
  resblock(34);   // d_r1
  resblock(42);   // d_r2

  // 9) d_dc1: conv-transpose 256->256, k4 s2 p1, 16->32 (WMMA, parity gather; M=16384)
  k_pack_w<<<4096, 256, 0, stream>>>(d_dc1w, WPK, 256, 256, 4, 4, 4096, 1);
  k_conv_wmma<<<dim3(256, 2), 256, 0, stream>>>(HB0, WPK, d_dc1b, BIGA,
      256, 16, 16, 256, 32, 32, 4, 4, 2, 1, 4096, 1);
  // 10) LN + ReLU -> f32 BIGB
  k_ln<<<16384, 256, 0, stream>>>(BIGA, d_n1g, d_n1b, nullptr, (float*)BIGB, nullptr, 1024);
  // 11) d_dc2 + tanh -> RECON
  k_deconv2_tanh<<<768, 256, 0, stream>>>((float*)BIGB, d_dc2w, d_dc2b, RECON);

  // 12) losses: out[0:16]=loss, [16:32]=recon, [32:48]=codebook, [48:64]=commit
  k_loss<<<16, 256, 0, stream>>>(RECON, XN, 12288, out + 16, nullptr);
  k_loss<<<16, 256, 0, stream>>>(S4, S3, 65536, out + 32, out + 48);
  k_finalize<<<1, 16, 0, stream>>>(out);
}
